// STFC_69475390980319
// MI455X (gfx1250) — compile-verified, matrix operand order checked by reading, not run on hardware
//
#include <hip/hip_runtime.h>

// ---------------------------------------------------------------------------
// Spiking-NN (PoissonEncoder -> Linear(784,10) -> LIF) over T=50 steps.
// Exact JAX threefry-2x32 RNG; matmul via v_wmma_f32_16x16x32_f16.
//
// - Each 128-thread (4-wave) workgroup owns batch rows [r0,r0+16) AND
//   [r0+4096,r0+4096+16): JAX uniform() pairs counters (n, n+N/2) in one
//   threefry block and N/2 == 4096*784, so one hash yields the uniforms for
//   (row,k) and (row+4096,k) -> RNG cost halved, each B fragment feeds 2 WMMAs.
// - The 25 K-chunks are split round-robin over the 4 waves; per-timestep
//   partial h fragments are reduced through a double-buffered LDS area with
//   ONE barrier per timestep (parity buffering removes the second barrier).
// - Bernoulli test done in integers: (bits>>9) < ceil(x * 2^23), exact.
// - 144 KB LDS/WG -> 2 WGs per 320 KB WGP -> 8 waves/WGP (2 per SIMD32),
//   enough to saturate VALU issue on this threefry-bound kernel.
// ---------------------------------------------------------------------------

typedef __attribute__((ext_vector_type(16))) _Float16 v16h;
typedef __attribute__((ext_vector_type(8)))  float    v8f;

#define HALF_N    3211264u               // (8192*784)/2 = 4096*784
#define KCH       25                     // ceil(784/32) K-chunks of 32
#define B_BYTES   (KCH*32*16*2)          // 25600 B of f16 W^T fragments
#define TH_WORDS  (KCH*32*16)            // 12800 u32 thresholds per row-group
#define P_OFF     (B_BYTES + 2*TH_WORDS*4)          // 128000
#define P_BYTES   (2*4*2*32*8*4)                    // parity*wave*group*lane*8 = 16384
#define SHMEM_BYTES (P_OFF + P_BYTES)               // 144384 B

__device__ __forceinline__ uint32_t rotl32(uint32_t x, uint32_t r) {
    return (x << r) | (x >> (32u - r));
}
__device__ __forceinline__ void tf_round(uint32_t& x0, uint32_t& x1, uint32_t r) {
    x0 += x1; x1 = rotl32(x1, r); x1 ^= x0;
}
// Exact JAX threefry2x32: 20 rounds, key injection every 4.
__device__ __forceinline__ void threefry2x32(uint32_t k0, uint32_t k1,
                                             uint32_t c0, uint32_t c1,
                                             uint32_t& o0, uint32_t& o1) {
    const uint32_t ks2 = 0x1BD11BDAu ^ k0 ^ k1;
    uint32_t x0 = c0 + k0, x1 = c1 + k1;
    tf_round(x0,x1,13); tf_round(x0,x1,15); tf_round(x0,x1,26); tf_round(x0,x1, 6);
    x0 += k1;  x1 += ks2 + 1u;
    tf_round(x0,x1,17); tf_round(x0,x1,29); tf_round(x0,x1,16); tf_round(x0,x1,24);
    x0 += ks2; x1 += k0 + 2u;
    tf_round(x0,x1,13); tf_round(x0,x1,15); tf_round(x0,x1,26); tf_round(x0,x1, 6);
    x0 += k0;  x1 += k1 + 3u;
    tf_round(x0,x1,17); tf_round(x0,x1,29); tf_round(x0,x1,16); tf_round(x0,x1,24);
    x0 += k1;  x1 += ks2 + 4u;
    tf_round(x0,x1,13); tf_round(x0,x1,15); tf_round(x0,x1,26); tf_round(x0,x1, 6);
    x0 += ks2; x1 += k0 + 5u;
    o0 = x0; o1 = x1;
}

__global__ void __launch_bounds__(128)
snn_lif_wmma(const float* __restrict__ x,   // [8192, 784]
             const float* __restrict__ W,   // [10, 784]
             const int*   __restrict__ Tp,  // scalar T
             float*       __restrict__ out) // [8192, 10]
{
    extern __shared__ char smem[];
    _Float16* sB  = (_Float16*)smem;                 // W^T fragments (frag layout)
    uint32_t* sTh = (uint32_t*)(smem + B_BYTES);     // ceil(x*2^23) thresholds, 2 groups
    float*    sP  = (float*)(smem + P_OFF);          // partial-h reduction buffers

    const int tid  = (int)threadIdx.x;
    const int wid  = tid >> 5;             // wave id 0..3 (K-split)
    const int lane = tid & 31;
    const int M    = lane & 15;            // A-frag row / C-frag column index
    const int hi   = (lane >> 4) & 1;      // lane group selects K sub-range
    const uint32_t r0   = (uint32_t)blockIdx.x * 16u;
    const uint32_t rowA = r0 + (uint32_t)M;

    // -------- one-time staging, chunks split across the 4 waves.
    // f16 16x32 A layout: lane<16 holds M=lane, K={2v,2v+1}(v<4),{16+2(v-4),..}(v>=4);
    // lanes>=16 add +8 to K. B loaded via the transposed-identical layout (N=lane&15).
    for (int c = wid; c < KCH; c += 4) {
        #pragma unroll
        for (int v = 0; v < 8; ++v) {
            const int kv = (v < 4 ? 2*v : 16 + 2*(v-4)) + (hi ? 8 : 0);
            #pragma unroll
            for (int e = 0; e < 2; ++e) {
                const int j = 2*v + e;
                const uint32_t k = (uint32_t)(c*32 + kv + e);
                _Float16 bv = (_Float16)0.0f;
                if (M < 10 && k < 784u) bv = (_Float16)W[(uint32_t)M*784u + k];
                sB[c*512 + lane*16 + j] = bv;
                uint32_t th0 = 0u, th1 = 0u;   // pad: m < 0 never fires
                if (k < 784u) {
                    // exact: u=m/2^23 < x  <=>  m < ceil(x*2^23)  (x*2^23, ceil exact in f32)
                    th0 = (uint32_t)ceilf(x[rowA*784u + k]          * 8388608.0f);
                    th1 = (uint32_t)ceilf(x[(rowA + 4096u)*784u + k] * 8388608.0f);
                }
                sTh[           c*512 + lane*16 + j] = th0;
                sTh[TH_WORDS + c*512 + lane*16 + j] = th1;
            }
        }
    }
    __syncthreads();

    const int T = *Tp;
    v8f vm0 = {}, vm1 = {}, cnt0 = {}, cnt1 = {};

    #pragma unroll 1
    for (int t = 0; t < T; ++t) {
        // key_t = fold_in(key(42), t) = threefry([0,42],[0,t]); wave-uniform -> SALU
        uint32_t kt0, kt1;
        threefry2x32(0u, 42u, 0u, (uint32_t)t, kt0, kt1);

        v8f h0 = {}, h1 = {};
        #pragma unroll 1
        for (int c = wid; c < KCH; c += 4) {
            const v16h bfrag = *(const v16h*)(sB + c*512 + lane*16);
            const uint32_t* tp0 = sTh +            c*512 + lane*16;
            const uint32_t* tp1 = sTh + TH_WORDS + c*512 + lane*16;

            v16h a0, a1;
            #pragma unroll
            for (int v = 0; v < 8; ++v) {
                const int kv = (v < 4 ? 2*v : 16 + 2*(v-4)) + (hi ? 8 : 0);
                #pragma unroll
                for (int e = 0; e < 2; ++e) {
                    const int j = 2*v + e;
                    const uint32_t k = (uint32_t)(c*32 + kv + e);
                    const uint32_t n = rowA*784u + k;   // flat index, first half
                    uint32_t b0, b1;
                    threefry2x32(kt0, kt1, n, n + HALF_N, b0, b1);
                    a0[j] = ((b0 >> 9) < tp0[j]) ? (_Float16)1.0f : (_Float16)0.0f;
                    a1[j] = ((b1 >> 9) < tp1[j]) ? (_Float16)1.0f : (_Float16)0.0f;
                }
            }
            h0 = __builtin_amdgcn_wmma_f32_16x16x32_f16(false, a0, false, bfrag,
                                                        (short)0, h0, false, false);
            h1 = __builtin_amdgcn_wmma_f32_16x16x32_f16(false, a1, false, bfrag,
                                                        (short)0, h1, false, false);
        }

        // ---- cross-wave K reduction; double-buffered, one barrier per step.
        const int par = t & 1;
        float* pw = sP + (size_t)((par*4 + wid)*2)*256 + lane*8;
        #pragma unroll
        for (int j = 0; j < 8; ++j) { pw[j] = h0[j]; pw[256 + j] = h1[j]; }
        __syncthreads();

        v8f H0 = {}, H1 = {};
        #pragma unroll
        for (int w = 0; w < 4; ++w) {
            const float* pr = sP + (size_t)((par*4 + w)*2)*256 + lane*8;
            #pragma unroll
            for (int j = 0; j < 8; ++j) { H0[j] += pr[j]; H1[j] += pr[256 + j]; }
        }

        // LIF: v += (h - v)/2 ; spike = (v >= 1) ; hard reset (redundant per wave)
        #pragma unroll
        for (int j = 0; j < 8; ++j) {
            const float nv0 = 0.5f * (vm0[j] + H0[j]);
            const float nv1 = 0.5f * (vm1[j] + H1[j]);
            const float s0 = (nv0 >= 1.0f) ? 1.0f : 0.0f;
            const float s1 = (nv1 >= 1.0f) ? 1.0f : 0.0f;
            cnt0[j] += s0;  cnt1[j] += s1;
            vm0[j] = nv0 * (1.0f - s0);
            vm1[j] = nv1 * (1.0f - s1);
        }
    }

    // C/D layout: VGPR j -> row (j + 8*hi), lane%16 -> column N. Wave 0 writes.
    const float inv = 1.0f / (float)T;
    if (wid == 0 && M < 10) {
        #pragma unroll
        for (int j = 0; j < 8; ++j) {
            const uint32_t m = (uint32_t)(j + (hi ? 8 : 0));
            out[(r0 + m)*10u + (uint32_t)M]         = cnt0[j] * inv;
            out[(r0 + 4096u + m)*10u + (uint32_t)M] = cnt1[j] * inv;
        }
    }
}

extern "C" void kernel_launch(void* const* d_in, const int* in_sizes, int n_in,
                              void* d_out, int out_size, void* d_ws, size_t ws_size,
                              hipStream_t stream) {
    const float* x  = (const float*)d_in[0];   // [8192,28,28]
    const float* W  = (const float*)d_in[1];   // [10,784]
    const int*   Tp = (const int*)d_in[2];     // T (scalar on device)
    float* out = (float*)d_out;                // [8192,10]

    (void)in_sizes; (void)n_in; (void)out_size; (void)d_ws; (void)ws_size;

    // 144 KB dynamic LDS per WG (gfx1250 WGP has 320 KB -> 2 WGs/WGP).
    hipFuncSetAttribute((const void*)snn_lif_wmma,
                        hipFuncAttributeMaxDynamicSharedMemorySize, SHMEM_BYTES);

    // 4096/16 = 256 tile-pairs; each 4-wave WG covers 16 low + 16 high rows.
    snn_lif_wmma<<<dim3(256), dim3(128), SHMEM_BYTES, stream>>>(x, W, Tp, out);
}
